// TSSA_Attention_NonCausal_1245540516156
// MI455X (gfx1250) — matrix operand order repaired and checked by reading, not compiled, not run on hardware
//
#include <hip/hip_runtime.h>

// ---------- problem constants ----------
constexpr int B_ = 4;
constexpr int T_ = 4096;
constexpr int C_ = 1024;
constexpr int H_ = 16;
constexpr int M_ = B_ * T_;   // 16384 rows
constexpr int K_ = C_;        // 1024
constexpr int N_ = C_;        // 1024

// ---------- types ----------
typedef __bf16 bf16;
typedef __attribute__((ext_vector_type(16))) __bf16 v16bf;
typedef __attribute__((ext_vector_type(8)))  __bf16 v8bf;
typedef __attribute__((ext_vector_type(4)))  __bf16 v4bf;
typedef __attribute__((ext_vector_type(8)))  float  v8f;
typedef __attribute__((ext_vector_type(4)))  int    v4i;

// ---------- async global->LDS availability (device pass only) ----------
#if defined(__gfx1250__) && \
    __has_builtin(__builtin_amdgcn_global_load_async_to_lds_b128) && \
    __has_builtin(__builtin_amdgcn_s_wait_asynccnt)
#define USE_ASYNC 1
// builtin expects (int4 addrspace(1)*, int4 addrspace(3)*, imm offset, imm cpol)
#define ASYNC_CP128(gsrc, ldst)                                   \
    __builtin_amdgcn_global_load_async_to_lds_b128(               \
        (__attribute__((address_space(1))) v4i*)(gsrc),           \
        (__attribute__((address_space(3))) v4i*)(ldst), 0, 0)
#else
#define USE_ASYNC 0
#endif

// ---------- fp32 -> bf16 convert (4-wide) ----------
__global__ void cvt_bf16_kernel(const float* __restrict__ in, bf16* __restrict__ out, int n) {
    int i = (blockIdx.x * 256 + threadIdx.x) * 4;
    if (i + 3 < n) {
        float4 f = *(const float4*)(in + i);
        v4bf o;
        o[0] = (bf16)f.x; o[1] = (bf16)f.y; o[2] = (bf16)f.z; o[3] = (bf16)f.w;
        *(v4bf*)(out + i) = o;
    }
}

// ---------- bf16 WMMA GEMM: C[M,N] = A[M,K] * Bw[N,K]^T + bias[N] ----------
// block tile 128x128, K-step 32, 256 threads = 8 waves (4x2), wave tile 32x64.
// LDS double-buffered; tiles staged with GLOBAL_LOAD_ASYNC_TO_LDS_B128.
// Wave tile kept at 32x64 so total VGPR use stays inside v0-v255 (no MSB churn).
constexpr int BM  = 128;
constexpr int BN  = 128;
constexpr int BK  = 32;
constexpr int LDK = 40;   // padded LDS row stride (bf16 elems); 80B rows keep 16B align

__global__ __launch_bounds__(256)
void gemm_bf16_kernel(const bf16* __restrict__ A, const bf16* __restrict__ Bw,
                      const float* __restrict__ bias, float* __restrict__ C)
{
    __shared__ bf16 As[2][BM * LDK];   // 2 * 10240 B
    __shared__ bf16 Bs[2][BN * LDK];   // 2 * 10240 B

    const int tid   = threadIdx.x;
    const int lane  = tid & 31;
    const int wave  = tid >> 5;       // 0..7
    const int wm    = wave & 3;       // 4 row-wave groups of 32 rows
    const int wn    = wave >> 2;      // 2 col-wave groups of 64 cols
    const int l16   = lane & 15;
    const int lhalf = lane >> 4;      // 0 or 1

    const int blockN = blockIdx.x * BN;
    const int blockM = blockIdx.y * BM;

    // stage one 128x32 A tile + 128x32 B tile into buffer `buf`
    // (4 x 16B async chunks per thread: 2 for A, 2 for B)
    auto stage = [&](int buf, int k0) {
#pragma unroll
        for (int i = 0; i < 2; ++i) {                 // A: 512 chunks
            int id  = tid + i * 256;
            int row = id >> 2;                        // 0..127
            int col = (id & 3) * 8;                   // 0,8,16,24
            const bf16* g = A + (size_t)(blockM + row) * K_ + k0 + col;
            bf16*       l = &As[buf][row * LDK + col];
#if USE_ASYNC
            ASYNC_CP128(g, l);
#else
            *(v8bf*)l = *(const v8bf*)g;
#endif
        }
#pragma unroll
        for (int i = 0; i < 2; ++i) {                 // B: 512 chunks
            int id  = tid + i * 256;
            int row = id >> 2;                        // 0..127
            int col = (id & 3) * 8;
            const bf16* g = Bw + (size_t)(blockN + row) * K_ + k0 + col;
            bf16*       l = &Bs[buf][row * LDK + col];
#if USE_ASYNC
            ASYNC_CP128(g, l);
#else
            *(v8bf*)l = *(const v8bf*)g;
#endif
        }
    };

    v8f acc[2][4];
#pragma unroll
    for (int tm = 0; tm < 2; ++tm)
#pragma unroll
        for (int tn = 0; tn < 4; ++tn)
#pragma unroll
            for (int r = 0; r < 8; ++r) acc[tm][tn][r] = 0.0f;

    union Frag { v16bf v; v8bf h[2]; };

    constexpr int kTiles = K_ / BK;   // 32
    stage(0, 0);

    for (int it = 0; it < kTiles; ++it) {
        const int cur = it & 1;
        if (it + 1 < kTiles) {
            stage(cur ^ 1, (it + 1) * BK);
#if USE_ASYNC
            // leave only the 4 newest async copies (next tile) outstanding
            __builtin_amdgcn_s_wait_asynccnt(4);
#endif
        } else {
#if USE_ASYNC
            __builtin_amdgcn_s_wait_asynccnt(0);
#endif
        }
        __syncthreads();

        const bf16* __restrict__ Ab = As[cur];
        const bf16* __restrict__ Bb = Bs[cur];

        // A fragments: lane<16 -> K{0..7,16..23}, lane>=16 -> K{8..15,24..31}
        Frag af[2];
#pragma unroll
        for (int tm = 0; tm < 2; ++tm) {
            int m = wm * 32 + tm * 16 + l16;
            const bf16* p = Ab + m * LDK + lhalf * 8;
            af[tm].h[0] = *(const v8bf*)(p);
            af[tm].h[1] = *(const v8bf*)(p + 16);
        }
        // B fragments: lane<16 -> K{0..7,16..23}, lane>=16 -> K{8..15,24..31} (N-major LDS)
        Frag bfr[4];
#pragma unroll
        for (int tn = 0; tn < 4; ++tn) {
            int n = wn * 64 + tn * 16 + l16;
            const bf16* p = Bb + n * LDK + lhalf * 8;
            bfr[tn].h[0] = *(const v8bf*)(p);
            bfr[tn].h[1] = *(const v8bf*)(p + 16);
        }

#pragma unroll
        for (int tm = 0; tm < 2; ++tm)
#pragma unroll
            for (int tn = 0; tn < 4; ++tn)
                acc[tm][tn] = __builtin_amdgcn_wmma_f32_16x16x32_bf16(
                    false, af[tm].v, false, bfr[tn].v,
                    (short)0, acc[tm][tn], false, false);

        __syncthreads();
    }

    // epilogue: lane 0-15 => N=l16, M=r ; lane 16-31 => N=l16, M=8+r
#pragma unroll
    for (int tn = 0; tn < 4; ++tn) {
        int col = blockN + wn * 64 + tn * 16 + l16;
        float bv = bias[col];
#pragma unroll
        for (int tm = 0; tm < 2; ++tm) {
            int rowBase = blockM + wm * 32 + tm * 16 + lhalf * 8;
#pragma unroll
            for (int r = 0; r < 8; ++r)
                C[(size_t)(rowBase + r) * N_ + col] = acc[tm][tn][r] + bv;
        }
    }
}

// ---------- per-(b,c) column sum of squares over T -> 1/max(ss,1e-24) ----------
__global__ void colnorm_kernel(const float* __restrict__ w, float* __restrict__ invn) {
    __shared__ float sh[4][64];
    int b  = blockIdx.x >> 4;          // 16 column-groups per batch
    int cg = blockIdx.x & 15;
    int d  = threadIdx.x & 63;
    int tg = threadIdx.x >> 6;         // 0..3
    int c  = cg * 64 + d;
    const float* p = w + (size_t)b * T_ * C_ + c;
    float s = 0.f;
    for (int t = tg; t < T_; t += 4) {
        float v = p[(size_t)t * C_];
        s += v * v;
    }
    sh[tg][d] = s;
    __syncthreads();
    if (tg == 0) {
        float tot = sh[0][d] + sh[1][d] + sh[2][d] + sh[3][d];
        invn[b * C_ + c] = 1.0f / fmaxf(tot, 1e-24f);
    }
}

// ---------- tmp[b,h,t] = temp[h] * sum_d (w^2 * invn) ----------
__global__ void headsq_kernel(const float* __restrict__ w, const float* __restrict__ invn,
                              const float* __restrict__ temp, float* __restrict__ tmpA) {
    int idx = blockIdx.x * 256 + threadIdx.x;    // B*H*T = 262144
    int t = idx & (T_ - 1);
    int h = (idx >> 12) & (H_ - 1);
    int b = idx >> 16;
    const float* p = w    + (size_t)(b * T_ + t) * C_ + h * 64;
    const float* q = invn + b * C_ + h * 64;
    float s = 0.f;
#pragma unroll
    for (int d = 0; d < 64; d += 4) {
        float4 wv = *(const float4*)(p + d);
        float4 iv = *(const float4*)(q + d);
        s += wv.x * wv.x * iv.x + wv.y * wv.y * iv.y
           + wv.z * wv.z * iv.z + wv.w * wv.w * iv.w;
    }
    tmpA[idx] = s * temp[h];
}

// ---------- softmax over the 16 heads per (b,t) ----------
__global__ void headsoftmax_kernel(const float* __restrict__ tmpA, float* __restrict__ Pi) {
    int idx = blockIdx.x * 256 + threadIdx.x;    // B*T = 16384
    int b = idx >> 12;
    int t = idx & (T_ - 1);
    const float* p = tmpA + (size_t)b * H_ * T_ + t;
    float v[H_];
    float mx = -3.4e38f;
#pragma unroll
    for (int h = 0; h < H_; ++h) { v[h] = p[(size_t)h * T_]; mx = fmaxf(mx, v[h]); }
    float den = 0.f;
#pragma unroll
    for (int h = 0; h < H_; ++h) { v[h] = __expf(v[h] - mx); den += v[h]; }
    float inv = 1.0f / den;
    float* o = Pi + (size_t)b * H_ * T_ + t;
#pragma unroll
    for (int h = 0; h < H_; ++h) o[(size_t)h * T_] = v[h] * inv;
}

// ---------- per-(b,h): pisum, dots[d], attn[d] = 1/(1 + dots/(pisum+1e-8)) ----------
__global__ void attn_kernel(const float* __restrict__ w, const float* __restrict__ Pi,
                            float* __restrict__ attnA) {
    __shared__ float shd[4][64];
    __shared__ float shp[4];
    int bh = blockIdx.x;               // B*H = 64
    int b = bh >> 4, h = bh & 15;
    int d  = threadIdx.x & 63;
    int tg = threadIdx.x >> 6;
    const float* pw = w  + (size_t)b * T_ * C_ + h * 64 + d;
    const float* pp = Pi + (size_t)bh * T_;
    float dd = 0.f, ps = 0.f;
    for (int t = tg; t < T_; t += 4) {
        float pi = pp[t];
        float wv = pw[(size_t)t * C_];
        dd += pi * wv * wv;
        ps += pi;
    }
    shd[tg][d] = dd;
    if (d == 0) shp[tg] = ps;
    __syncthreads();
    if (tg == 0) {
        float dtot  = shd[0][d] + shd[1][d] + shd[2][d] + shd[3][d];
        float pisum = shp[0] + shp[1] + shp[2] + shp[3];
        float dots  = dtot / (pisum + 1e-8f);
        attnA[bh * 64 + d] = 1.0f / (1.0f + dots);
    }
}

// ---------- y = -(w * Pi) * attn, output bf16 (4-wide) ----------
__global__ void ymake_kernel(const float* __restrict__ w, const float* __restrict__ Pi,
                             const float* __restrict__ attnA, bf16* __restrict__ y) {
    int i = (blockIdx.x * 256 + threadIdx.x) * 4;   // B*T*C elements
    int c = i & (C_ - 1);
    int t = (i >> 10) & (T_ - 1);
    int b = i >> 22;
    int h = c >> 6, d = c & 63;
    float4 wv = *(const float4*)(w + i);
    float  pi = Pi[(size_t)(b * H_ + h) * T_ + t];
    float4 av = *(const float4*)(attnA + (b * H_ + h) * 64 + d);
    v4bf o;
    o[0] = (bf16)(-wv.x * pi * av.x);
    o[1] = (bf16)(-wv.y * pi * av.y);
    o[2] = (bf16)(-wv.z * pi * av.z);
    o[3] = (bf16)(-wv.w * pi * av.w);
    *(v4bf*)(y + i) = o;
}

// ---------- launch ----------
extern "C" void kernel_launch(void* const* d_in, const int* in_sizes, int n_in,
                              void* d_out, int out_size, void* d_ws, size_t ws_size,
                              hipStream_t stream) {
    const float* x    = (const float*)d_in[0];
    const float* Wa   = (const float*)d_in[1];
    const float* ba   = (const float*)d_in[2];
    const float* Wp   = (const float*)d_in[3];
    const float* bp   = (const float*)d_in[4];
    const float* temp = (const float*)d_in[5];
    float* out = (float*)d_out;

    char* ws = (char*)d_ws;
    // x_bf reused as y_bf after GEMM1 consumes x.
    bf16*  x_bf  = (bf16*)(ws + 0);             // 33,554,432 B
    bf16*  y_bf  = (bf16*)(ws + 0);
    bf16*  wa_bf = (bf16*)(ws + 33554432);      //  2,097,152 B
    bf16*  wp_bf = (bf16*)(ws + 35651584);      //  2,097,152 B
    float* w     = (float*)(ws + 37748736);     // 67,108,864 B
    float* invn  = (float*)(ws + 104857600);    //     16,384 B
    float* tmpA  = (float*)(ws + 104873984);    //  1,048,576 B
    float* Pi    = (float*)(ws + 105922560);    //  1,048,576 B
    float* attnA = (float*)(ws + 106971136);    //     16,384 B
    // total ~102 MB

    cvt_bf16_kernel<<<16384, 256, 0, stream>>>(x,  x_bf,  M_ * K_);
    cvt_bf16_kernel<<<1024,  256, 0, stream>>>(Wa, wa_bf, N_ * K_);
    cvt_bf16_kernel<<<1024,  256, 0, stream>>>(Wp, wp_bf, N_ * K_);

    gemm_bf16_kernel<<<dim3(N_ / BN, M_ / BM), 256, 0, stream>>>(x_bf, wa_bf, ba, w);

    colnorm_kernel    <<<B_ * (C_ / 64), 256, 0, stream>>>(w, invn);
    headsq_kernel     <<<(B_ * H_ * T_) / 256, 256, 0, stream>>>(w, invn, temp, tmpA);
    headsoftmax_kernel<<<(B_ * T_) / 256, 256, 0, stream>>>(tmpA, Pi);
    attn_kernel       <<<B_ * H_, 256, 0, stream>>>(w, Pi, attnA);
    ymake_kernel      <<<(B_ * T_ * C_) / 1024, 256, 0, stream>>>(w, Pi, attnA, y_bf);

    gemm_bf16_kernel<<<dim3(N_ / BN, M_ / BM), 256, 0, stream>>>(y_bf, wp_bf, bp, out);
}
